// TieredMoELayer_32238024524299
// MI455X (gfx1250) — matrix operand order repaired and checked by reading, not compiled, not run on hardware
//
#include <hip/hip_runtime.h>
#include <hip/hip_bf16.h>

// ---- problem constants (match reference) ----
#define T_TOK   4096      // B*S
#define DDIM    1024
#define HDIM    2048
#define NEXP    8
#define TOPK    2
#define LN_EPS  1e-5f

typedef __bf16 bf16_t;
typedef __attribute__((ext_vector_type(16))) __bf16 v16bf;
typedef __attribute__((ext_vector_type(8)))  float  v8f;

struct Pack32 { uint4 lo; uint4 hi; };   // 32 bytes == v16bf

// ---------------------------------------------------------------------------
// WMMA fragment loaders (wave32; contiguous k = 16h + s order, same in A and B, not the ISA table order).
// A 16x32 bf16 (MxK): lane m = l&15; lanes<16 hold K 0..15, lanes>=16 hold
//   K 16..31 (same k order as B)
// -> vector elements 0..15 = K[half..half+15]:
//    one contiguous 32-byte chunk of the row-major row.
// ---------------------------------------------------------------------------
__device__ inline v16bf load_frag_a(const bf16_t* __restrict__ row, int lane) {
  const int half = (lane >= 16) ? 16 : 0;
  const uint4* p = reinterpret_cast<const uint4*>(row + half);
  Pack32 pk;
  pk.lo = p[0];   // K half .. half+7
  pk.hi = p[1];   // K half+8 .. half+15
  return __builtin_bit_cast(v16bf, pk);
}

// B 32x16 bf16 (KxN): lane n = l&15; lanes<16 hold K 0..15, lanes>=16 hold
// K 16..31; VGPR i packs (2i,2i+1) -> elements 0..15 = K[off..off+15]:
// one contiguous 32-byte chunk of the (pre-converted bf16) weight row.
__device__ inline v16bf load_frag_b(const bf16_t* __restrict__ wrow, int lane) {
  const int off = (lane >= 16) ? 16 : 0;
  const uint4* p = reinterpret_cast<const uint4*>(wrow + off);
  Pack32 pk;
  pk.lo = p[0];
  pk.hi = p[1];
  return __builtin_bit_cast(v16bf, pk);
}

__device__ inline v8f wmma_bf16(v16bf a, v16bf b, v8f c) {
  return __builtin_amdgcn_wmma_f32_16x16x32_bf16(false, a, false, b, (short)0, c,
                                                 false, false);
}

// ---------------------------------------------------------------------------
// 0) f32 -> bf16 weight pre-conversion (once per launch; streams HBM once)
// ---------------------------------------------------------------------------
__global__ __launch_bounds__(256) void moe_cvt_bf16(
    const float* __restrict__ src, bf16_t* __restrict__ dst, int n) {
  const int i0 = (blockIdx.x * 256 + threadIdx.x) * 8;
  if (i0 >= n) return;
  const float4 a = *reinterpret_cast<const float4*>(src + i0);
  const float4 b = *reinterpret_cast<const float4*>(src + i0 + 4);
  bf16_t o[8] = {(bf16_t)a.x, (bf16_t)a.y, (bf16_t)a.z, (bf16_t)a.w,
                 (bf16_t)b.x, (bf16_t)b.y, (bf16_t)b.z, (bf16_t)b.w};
  *reinterpret_cast<uint4*>(dst + i0) = __builtin_bit_cast(uint4, o);
}

// ---------------------------------------------------------------------------
// 1) LayerNorm + router (one block per token)
// ---------------------------------------------------------------------------
__global__ __launch_bounds__(256) void moe_ln_router(
    const float* __restrict__ x, const float* __restrict__ ln_w,
    const float* __restrict__ ln_b, const float* __restrict__ Wr,
    bf16_t* __restrict__ xn, float* __restrict__ gateW, int* __restrict__ topIdx,
    int* __restrict__ counts, float* __restrict__ probs) {
  const int t = blockIdx.x;
  const int tid = threadIdx.x;
  __shared__ float s_red[256];
  __shared__ float s_log[NEXP * 256];

  const int d0 = tid * 4;
  float xv[4];
#pragma unroll
  for (int j = 0; j < 4; ++j) xv[j] = x[(size_t)t * DDIM + d0 + j];

  float s = xv[0] + xv[1] + xv[2] + xv[3];
  s_red[tid] = s; __syncthreads();
  for (int st = 128; st > 0; st >>= 1) {
    if (tid < st) s_red[tid] += s_red[tid + st];
    __syncthreads();
  }
  const float mu = s_red[0] * (1.0f / DDIM);
  __syncthreads();
  float v = 0.f;
#pragma unroll
  for (int j = 0; j < 4; ++j) { float d = xv[j] - mu; v += d * d; }
  s_red[tid] = v; __syncthreads();
  for (int st = 128; st > 0; st >>= 1) {
    if (tid < st) s_red[tid] += s_red[tid + st];
    __syncthreads();
  }
  const float rstd = rsqrtf(s_red[0] * (1.0f / DDIM) + LN_EPS);
  __syncthreads();

  float pl[NEXP];
#pragma unroll
  for (int e = 0; e < NEXP; ++e) pl[e] = 0.f;
#pragma unroll
  for (int j = 0; j < 4; ++j) {
    const int d = d0 + j;
    const float xnv = (xv[j] - mu) * rstd * ln_w[d] + ln_b[d];
    xn[(size_t)t * DDIM + d] = (bf16_t)xnv;
#pragma unroll
    for (int e = 0; e < NEXP; ++e) pl[e] += xnv * Wr[(size_t)e * DDIM + d];
  }
#pragma unroll
  for (int e = 0; e < NEXP; ++e) s_log[e * 256 + tid] = pl[e];
  __syncthreads();
  if (tid < NEXP) {                  // fixed-order column sum (deterministic)
    float acc = 0.f;
    for (int i = 0; i < 256; ++i) acc += s_log[tid * 256 + i];
    s_red[tid] = acc;
  }
  __syncthreads();
  if (tid == 0) {
    float lg[NEXP];
#pragma unroll
    for (int e = 0; e < NEXP; ++e) lg[e] = s_red[e];
    float v0 = -INFINITY; int i0 = 0;
#pragma unroll
    for (int e = 0; e < NEXP; ++e) if (lg[e] > v0) { v0 = lg[e]; i0 = e; }
    float v1 = -INFINITY; int i1 = 0;
#pragma unroll
    for (int e = 0; e < NEXP; ++e)
      if (e != i0 && lg[e] > v1) { v1 = lg[e]; i1 = e; }
    const float e1 = __expf(v1 - v0);
    const float inv = 1.0f / (1.0f + e1);
    gateW[t * 2 + 0] = inv;
    gateW[t * 2 + 1] = e1 * inv;
    topIdx[t * 2 + 0] = i0;
    topIdx[t * 2 + 1] = i1;
    atomicAdd(&counts[i0], 1);
    atomicAdd(&counts[i1], 1);
    float m8 = -INFINITY;
#pragma unroll
    for (int e = 0; e < NEXP; ++e) m8 = fmaxf(m8, lg[e]);
    float den = 0.f;
#pragma unroll
    for (int e = 0; e < NEXP; ++e) den += __expf(lg[e] - m8);
    const float rden = 1.0f / den;
#pragma unroll
    for (int e = 0; e < NEXP; ++e)
      probs[(size_t)t * NEXP + e] = __expf(lg[e] - m8) * rden;
  }
}

// ---------------------------------------------------------------------------
// 2) offsets (prefix sum over 8 experts) + cursor reset
// ---------------------------------------------------------------------------
__global__ void moe_offsets(const int* __restrict__ counts, int* __restrict__ offs,
                            int* __restrict__ cursor) {
  if (threadIdx.x == 0) {
    int acc = 0;
    for (int e = 0; e < NEXP; ++e) { offs[e] = acc; acc += counts[e]; cursor[e] = 0; }
    offs[NEXP] = acc;
  }
}

// 3) scatter tokens into per-expert lists (list entry = t*2 + slot)
__global__ __launch_bounds__(256) void moe_scatter(
    const int* __restrict__ topIdx, const int* __restrict__ offs,
    int* __restrict__ cursor, int* __restrict__ list) {
  const int t = blockIdx.x * 256 + threadIdx.x;
  if (t >= T_TOK) return;
#pragma unroll
  for (int k = 0; k < TOPK; ++k) {
    const int e = topIdx[t * 2 + k];
    const int p = atomicAdd(&cursor[e], 1);
    list[offs[e] + p] = t * 2 + k;
  }
}

// ---------------------------------------------------------------------------
// 4) expert FFN stage 1: h = relu(xn @ W1[e]^T + b1[e])  -> bf16 hbuf
//    grid: (4 col-groups of 512, 128 token-tiles of 32, 8 experts)
//    8 waves/block; each wave: 32 rows x 64 cols, K-loop over D
// ---------------------------------------------------------------------------
__global__ __launch_bounds__(256) void moe_ffn1(
    const bf16_t* __restrict__ xn, const bf16_t* __restrict__ W1b,
    const float* __restrict__ b1, const int* __restrict__ list,
    const int* __restrict__ offs, const int* __restrict__ counts,
    bf16_t* __restrict__ hbuf) {
  const int e = blockIdx.z;
  const int cnt = counts[e];
  const int pbase = blockIdx.y * 32;
  if (pbase >= cnt) return;                       // block-uniform exit
  const int lane = threadIdx.x & 31;
  const int wave = threadIdx.x >> 5;
  const int nbase = blockIdx.x * 512 + wave * 64;
  const int base = offs[e];
  const int m = lane & 15;

  int pa0 = pbase + m;       if (pa0 >= cnt) pa0 = cnt - 1;
  int pa1 = pbase + 16 + m;  if (pa1 >= cnt) pa1 = cnt - 1;
  const bf16_t* arow0 = xn + (size_t)(list[base + pa0] >> 1) * DDIM;
  const bf16_t* arow1 = xn + (size_t)(list[base + pa1] >> 1) * DDIM;
  const bf16_t* wrow = W1b + ((size_t)e * HDIM + (nbase + m)) * DDIM;

  v8f acc[2][4];
#pragma unroll
  for (int h = 0; h < 2; ++h)
#pragma unroll
    for (int j = 0; j < 4; ++j)
      acc[h][j] = (v8f){0.f,0.f,0.f,0.f,0.f,0.f,0.f,0.f};

  for (int kt = 0; kt < DDIM; kt += 32) {
    if (kt + 32 < DDIM) {
      __builtin_prefetch(arow0 + kt + 32, 0, 0);   // global_prefetch_b8
      __builtin_prefetch(wrow + kt + 32, 0, 0);
    }
    const v16bf a0 = load_frag_a(arow0 + kt, lane);
    const v16bf a1 = load_frag_a(arow1 + kt, lane);
#pragma unroll
    for (int j = 0; j < 4; ++j) {
      const v16bf b = load_frag_b(wrow + (size_t)j * 16 * DDIM + kt, lane);
      acc[0][j] = wmma_bf16(a0, b, acc[0][j]);
      acc[1][j] = wmma_bf16(a1, b, acc[1][j]);
    }
  }

  const int n = lane & 15;
  const int mhalf = (lane >= 16) ? 8 : 0;
#pragma unroll
  for (int h = 0; h < 2; ++h) {
#pragma unroll
    for (int j = 0; j < 4; ++j) {
      const int col = nbase + j * 16 + n;
      const float bias = b1[e * HDIM + col];
#pragma unroll
      for (int r = 0; r < 8; ++r) {
        const int pp = pbase + h * 16 + mhalf + r;
        if (pp < cnt) {
          float vv = acc[h][j][r] + bias;
          vv = vv > 0.f ? vv : 0.f;
          hbuf[(size_t)(base + pp) * HDIM + col] = (bf16_t)vv;
        }
      }
    }
  }
}

// ---------------------------------------------------------------------------
// 5) expert FFN stage 2: y = h @ W2[e]^T + b2[e] -> ycontrib[(t*2+k), D]
//    grid: (2 col-groups of 512, 128 pos-tiles of 32, 8 experts)
// ---------------------------------------------------------------------------
__global__ __launch_bounds__(256) void moe_ffn2(
    const bf16_t* __restrict__ hbuf, const bf16_t* __restrict__ W2b,
    const float* __restrict__ b2, const int* __restrict__ list,
    const int* __restrict__ offs, const int* __restrict__ counts,
    float* __restrict__ ycontrib) {
  const int e = blockIdx.z;
  const int cnt = counts[e];
  const int pbase = blockIdx.y * 32;
  if (pbase >= cnt) return;
  const int lane = threadIdx.x & 31;
  const int wave = threadIdx.x >> 5;
  const int nbase = blockIdx.x * 512 + wave * 64;
  const int base = offs[e];
  const int m = lane & 15;

  int pa0 = pbase + m;       if (pa0 >= cnt) pa0 = cnt - 1;
  int pa1 = pbase + 16 + m;  if (pa1 >= cnt) pa1 = cnt - 1;
  const bf16_t* arow0 = hbuf + (size_t)(base + pa0) * HDIM;
  const bf16_t* arow1 = hbuf + (size_t)(base + pa1) * HDIM;
  const bf16_t* wrow = W2b + ((size_t)e * DDIM + (nbase + m)) * HDIM;

  v8f acc[2][4];
#pragma unroll
  for (int h = 0; h < 2; ++h)
#pragma unroll
    for (int j = 0; j < 4; ++j)
      acc[h][j] = (v8f){0.f,0.f,0.f,0.f,0.f,0.f,0.f,0.f};

  for (int kt = 0; kt < HDIM; kt += 32) {
    if (kt + 32 < HDIM) {
      __builtin_prefetch(arow0 + kt + 32, 0, 0);
      __builtin_prefetch(wrow + kt + 32, 0, 0);
    }
    const v16bf a0 = load_frag_a(arow0 + kt, lane);
    const v16bf a1 = load_frag_a(arow1 + kt, lane);
#pragma unroll
    for (int j = 0; j < 4; ++j) {
      const v16bf b = load_frag_b(wrow + (size_t)j * 16 * HDIM + kt, lane);
      acc[0][j] = wmma_bf16(a0, b, acc[0][j]);
      acc[1][j] = wmma_bf16(a1, b, acc[1][j]);
    }
  }

  const int n = lane & 15;
  const int mhalf = (lane >= 16) ? 8 : 0;
#pragma unroll
  for (int h = 0; h < 2; ++h) {
#pragma unroll
    for (int j = 0; j < 4; ++j) {
      const int col = nbase + j * 16 + n;
      const float bias = b2[e * DDIM + col];
#pragma unroll
      for (int r = 0; r < 8; ++r) {
        const int pp = pbase + h * 16 + mhalf + r;
        if (pp < cnt) {
          const int dest = list[base + pp];            // t*2 + slot
          ycontrib[(size_t)dest * DDIM + col] = acc[h][j][r] + bias;
        }
      }
    }
  }
}

// ---------------------------------------------------------------------------
// 6) combine: out = x + g0*yc0 + g1*yc1   (4 floats/thread)
// ---------------------------------------------------------------------------
__global__ __launch_bounds__(256) void moe_combine(
    const float* __restrict__ x, const float* __restrict__ gateW,
    const float* __restrict__ ycontrib, float* __restrict__ out) {
  const size_t gid = (size_t)blockIdx.x * 256 + threadIdx.x;
  const size_t i0 = gid * 4;
  const int t = (int)(i0 >> 10);          // DDIM = 1024
  const int d = (int)(i0 & (DDIM - 1));
  const float g0 = gateW[t * 2 + 0];
  const float g1 = gateW[t * 2 + 1];
  const float* y0 = ycontrib + (size_t)(t * 2 + 0) * DDIM + d;
  const float* y1 = ycontrib + (size_t)(t * 2 + 1) * DDIM + d;
#pragma unroll
  for (int j = 0; j < 4; ++j)
    out[i0 + j] = x[i0 + j] + g0 * y0[j] + g1 * y1[j];
}

// 7) aux loss: single-block deterministic reduction
__global__ __launch_bounds__(256) void moe_aux(
    const float* __restrict__ probs, const int* __restrict__ counts,
    float* __restrict__ out_aux) {
  __shared__ float s[NEXP * 256];
  const int tid = threadIdx.x;
  float acc[NEXP];
#pragma unroll
  for (int e = 0; e < NEXP; ++e) acc[e] = 0.f;
  for (int t = tid; t < T_TOK; t += 256)
#pragma unroll
    for (int e = 0; e < NEXP; ++e) acc[e] += probs[(size_t)t * NEXP + e];
#pragma unroll
  for (int e = 0; e < NEXP; ++e) s[e * 256 + tid] = acc[e];
  __syncthreads();
  for (int st = 128; st > 0; st >>= 1) {
    if (tid < st)
#pragma unroll
      for (int e = 0; e < NEXP; ++e) s[e * 256 + tid] += s[e * 256 + tid + st];
    __syncthreads();
  }
  if (tid == 0) {
    float aux = 0.f;
    const float invT = 1.0f / (float)T_TOK;
#pragma unroll
    for (int e = 0; e < NEXP; ++e)
      aux += ((float)counts[e] * invT) * (s[e * 256] * invT);
    out_aux[0] = (float)NEXP * aux;
  }
}

// ---------------------------------------------------------------------------
extern "C" void kernel_launch(void* const* d_in, const int* in_sizes, int n_in,
                              void* d_out, int out_size, void* d_ws, size_t ws_size,
                              hipStream_t stream) {
  const float* x    = (const float*)d_in[0];
  const float* ln_w = (const float*)d_in[1];
  const float* ln_b = (const float*)d_in[2];
  const float* Wr   = (const float*)d_in[3];
  const float* W1   = (const float*)d_in[4];
  const float* b1   = (const float*)d_in[5];
  const float* W2   = (const float*)d_in[6];
  const float* b2   = (const float*)d_in[7];
  float* out = (float*)d_out;

  // workspace carve (aligned to 256B)
  char* w = (char*)d_ws;
  auto alloc = [&](size_t bytes) {
    char* p = w;
    w += (bytes + 255) & ~(size_t)255;
    return p;
  };
  const int nW1 = NEXP * HDIM * DDIM;   // 16,777,216
  const int nW2 = NEXP * DDIM * HDIM;   // 16,777,216
  bf16_t* xn    = (bf16_t*)alloc((size_t)T_TOK * DDIM * sizeof(bf16_t));
  bf16_t* hbuf  = (bf16_t*)alloc((size_t)T_TOK * TOPK * HDIM * sizeof(bf16_t));
  float*  yc    = (float*)alloc((size_t)T_TOK * TOPK * DDIM * sizeof(float));
  bf16_t* W1b   = (bf16_t*)alloc((size_t)nW1 * sizeof(bf16_t));
  bf16_t* W2b   = (bf16_t*)alloc((size_t)nW2 * sizeof(bf16_t));
  float*  probs = (float*)alloc((size_t)T_TOK * NEXP * sizeof(float));
  float*  gateW = (float*)alloc((size_t)T_TOK * TOPK * sizeof(float));
  int*   topIdx = (int*)alloc((size_t)T_TOK * TOPK * sizeof(int));
  int*   list   = (int*)alloc((size_t)T_TOK * TOPK * sizeof(int));
  int*   counts = (int*)alloc(NEXP * sizeof(int));
  int*   offs   = (int*)alloc((NEXP + 1) * sizeof(int));
  int*   cursor = (int*)alloc(NEXP * sizeof(int));

  hipMemsetAsync(counts, 0, NEXP * sizeof(int), stream);

  // pre-convert weights to bf16 (streams each tensor once)
  moe_cvt_bf16<<<nW1 / (256 * 8), 256, 0, stream>>>(W1, W1b, nW1);
  moe_cvt_bf16<<<nW2 / (256 * 8), 256, 0, stream>>>(W2, W2b, nW2);

  moe_ln_router<<<T_TOK, 256, 0, stream>>>(x, ln_w, ln_b, Wr, xn, gateW, topIdx,
                                           counts, probs);
  moe_offsets<<<1, 32, 0, stream>>>(counts, offs, cursor);
  moe_scatter<<<T_TOK / 256, 256, 0, stream>>>(topIdx, offs, cursor, list);

  moe_ffn1<<<dim3(HDIM / 512, T_TOK / 32, NEXP), 256, 0, stream>>>(
      xn, W1b, b1, list, offs, counts, hbuf);
  moe_ffn2<<<dim3(DDIM / 512, T_TOK / 32, NEXP), 256, 0, stream>>>(
      hbuf, W2b, b2, list, offs, counts, yc);

  moe_combine<<<(T_TOK * (size_t)DDIM) / 1024, 256, 0, stream>>>(x, gateW, yc, out);
  moe_aux<<<1, 256, 0, stream>>>(probs, counts, out + (size_t)T_TOK * DDIM);
}
